// Generator_80333068304701
// MI455X (gfx1250) — matrix operand order repaired and checked by reading, not compiled, not run on hardware
//
#include <hip/hip_runtime.h>
#include <math.h>

// ---------------- problem constants ----------------
static constexpr int NB    = 64;        // graphs
static constexpr int NN    = 384;       // nodes per graph
static constexpr int NOISE = 128;
static constexpr int CED   = 64;
static constexpr int INDIM = NOISE + CED;   // 192
static constexpr int HID   = 512;
static constexpr int FEAT  = 256;
static constexpr int ROWS  = NB * NN;       // 24576
static constexpr int PP    = NN * (NN - 1) / 2;   // 73536
static constexpr int KSEL  = 3677;          // round(0.05 * PP)
static constexpr int NBINS = 2048;

// d_out element offsets (all 4-byte elements)
static constexpr size_t X_ELE     = (size_t)ROWS * FEAT;          // f32
static constexpr size_t EDGE_ELE  = (size_t)2 * NB * 2 * KSEL;    // i32
static constexpr size_t BATCH_ELE = (size_t)ROWS;                 // i32
// top_probs follow: NB*KSEL f32

// ---------------- vector types ----------------
typedef __attribute__((ext_vector_type(16))) __bf16 v16bf;
typedef __attribute__((ext_vector_type(8)))  float  v8f;
typedef __attribute__((ext_vector_type(4)))  int    int4v;

__device__ __forceinline__ unsigned short f2bf(float f) {
  unsigned u = __float_as_uint(f);
  unsigned r = u + 0x7FFFu + ((u >> 16) & 1u);   // RNE
  return (unsigned short)(r >> 16);
}
__device__ __forceinline__ float bf2f(unsigned short h) {
  return __uint_as_float(((unsigned)h) << 16);
}

// ---------------- CDNA5 async global->LDS copy (ASYNCcnt path) ----------------
#if __has_builtin(__builtin_amdgcn_global_load_async_to_lds_b128)
#define ASYNC_LDS 1
#else
#define ASYNC_LDS 0
#endif

__device__ __forceinline__ void cp16(const unsigned short* g, unsigned short* l) {
#if ASYNC_LDS
  __builtin_amdgcn_global_load_async_to_lds_b128(
      (__attribute__((address_space(1))) int4v*)g,
      (__attribute__((address_space(3))) int4v*)l, 0, 0);
#else
  *(int4v*)l = *(const int4v*)g;
#endif
}

__device__ __forceinline__ void async_wait_all() {
#if ASYNC_LDS
#if __has_builtin(__builtin_amdgcn_s_wait_asynccnt)
  __builtin_amdgcn_s_wait_asynccnt(0);
#else
  asm volatile("s_wait_asynccnt 0x0" ::: "memory");
#endif
#endif
}

union FragU { int4v q[2]; v16bf v; };

// A fragment: lane holds K = kh*8+{0..7} and kh*8+16+{0..7} of its row ([m][k] tile, 32 k)
__device__ __forceinline__ v16bf frag_a(const unsigned short* p) {
  FragU u;
  u.q[0] = *(const int4v*)(p);
  u.q[1] = *(const int4v*)(p + 16);
  return u.v;
}
// B fragment: lane holds K = kh*8+{0..7} and kh*8+16+{0..7} of its column (staged as [n][k] rows)
__device__ __forceinline__ v16bf frag_b(const unsigned short* p) {
  FragU u;
  u.q[0] = *(const int4v*)(p);
  u.q[1] = *(const int4v*)(p + 16);
  return u.v;
}

// ---------------- bf16 conversion / packing kernels ----------------
__global__ void k_build_a0(const float* __restrict__ z, const int* __restrict__ labels,
                           const float* __restrict__ ET, unsigned short* __restrict__ A0) {
  int e = blockIdx.x * 256 + threadIdx.x;
  if (e >= ROWS * INDIM) return;
  int row = e / INDIM, c = e % INDIM;
  float val;
  if (c < NOISE) {
    val = z[(size_t)row * NOISE + c];
  } else {
    int b = row / NN;
    val = ET[labels[b] * CED + (c - NOISE)];
  }
  A0[e] = f2bf(val);
}

// W [Kd][Nd] f32  ->  WT [Nd][Kd] bf16
__global__ void k_wt(const float* __restrict__ W, unsigned short* __restrict__ WT,
                     int Kd, int Nd) {
  int e = blockIdx.x * 256 + threadIdx.x;
  if (e >= Kd * Nd) return;
  int k = e / Nd, n = e % Nd;
  WT[(size_t)n * Kd + k] = f2bf(W[e]);
}

// ---------------- WMMA GEMM:  out = act(A[M,K] @ Bt[N,K]^T + bias) ----------------
__global__ __launch_bounds__(256) void k_gemm(const unsigned short* __restrict__ A,
                                              const unsigned short* __restrict__ Bt,
                                              const float* __restrict__ bias,
                                              float* __restrict__ outF,
                                              unsigned short* __restrict__ outB,
                                              int Nc, int K, int relu) {
  __shared__ __align__(16) unsigned short As[2][128 * 32];
  __shared__ __align__(16) unsigned short Bs[2][64 * 32];
  const int t = threadIdx.x;
  const int m0 = blockIdx.y * 128;
  const int n0 = blockIdx.x * 64;
  const int lane = t & 31, wave = t >> 5;
  const int wm = wave >> 2, wn = wave & 3;
  const int mrow = lane & 15, kh = lane >> 4;

  // per-thread staging coordinates (16B chunks)
  const int ar0 = t >> 2, ar1 = ar0 + 64, seg = (t & 3) * 8;
  const int br = t >> 2;

  auto stage = [&](int buf, int k0) {
    cp16(A + (size_t)(m0 + ar0) * K + k0 + seg, &As[buf][ar0 * 32 + seg]);
    cp16(A + (size_t)(m0 + ar1) * K + k0 + seg, &As[buf][ar1 * 32 + seg]);
    cp16(Bt + (size_t)(n0 + br) * K + k0 + seg, &Bs[buf][br * 32 + seg]);
  };

  v8f acc[4];
#pragma unroll
  for (int s = 0; s < 4; ++s)
#pragma unroll
    for (int r = 0; r < 8; ++r) acc[s][r] = 0.0f;

  const int nsteps = K >> 5;
  stage(0, 0);
  for (int i = 0; i < nsteps; ++i) {
    const int cur = i & 1;
    async_wait_all();      // tile `cur` landed in LDS (this wave's copies)
    __syncthreads();       // all waves done copying `cur` / done reading `cur^1`
    if (i + 1 < nsteps) stage(cur ^ 1, (i + 1) << 5);   // overlap with compute

    const unsigned short* Asb = As[cur];
    const unsigned short* Bsb = Bs[cur];
    v16bf bfrag = frag_b(Bsb + (wn * 16 + mrow) * 32 + kh * 8);
    v16bf a0 = frag_a(Asb + (wm * 64 + 0 + mrow) * 32 + kh * 8);
    v16bf a1 = frag_a(Asb + (wm * 64 + 16 + mrow) * 32 + kh * 8);
    v16bf a2 = frag_a(Asb + (wm * 64 + 32 + mrow) * 32 + kh * 8);
    v16bf a3 = frag_a(Asb + (wm * 64 + 48 + mrow) * 32 + kh * 8);
    acc[0] = __builtin_amdgcn_wmma_f32_16x16x32_bf16(false, a0, false, bfrag, (short)0, acc[0], false, false);
    acc[1] = __builtin_amdgcn_wmma_f32_16x16x32_bf16(false, a1, false, bfrag, (short)0, acc[1], false, false);
    acc[2] = __builtin_amdgcn_wmma_f32_16x16x32_bf16(false, a2, false, bfrag, (short)0, acc[2], false, false);
    acc[3] = __builtin_amdgcn_wmma_f32_16x16x32_bf16(false, a3, false, bfrag, (short)0, acc[3], false, false);
  }

  const int col = n0 + wn * 16 + mrow;
  const float bv = bias ? bias[col] : 0.0f;
#pragma unroll
  for (int s = 0; s < 4; ++s) {
#pragma unroll
    for (int r = 0; r < 8; ++r) {
      int row = m0 + wm * 64 + s * 16 + r + kh * 8;
      float v = acc[s][r] + bv;
      if (relu) v = fmaxf(v, 0.0f);
      size_t o = (size_t)row * Nc + col;
      if (outF) outF[o] = v;
      if (outB) outB[o] = f2bf(v);
    }
  }
}

// ---------------- per-row sum of squares of H (bf16 [ROWS][HID]) ----------------
__global__ void k_rowsq(const unsigned short* __restrict__ H, float* __restrict__ sq) {
  int row = blockIdx.x * 8 + (threadIdx.x >> 5);
  int lane = threadIdx.x & 31;
  const unsigned short* r = H + (size_t)row * HID + lane * 16;
  float s = 0.0f;
#pragma unroll
  for (int c = 0; c < 2; ++c) {
    union { int4v q; unsigned short h[8]; } u;
    u.q = *(const int4v*)(r + c * 8);
#pragma unroll
    for (int j = 0; j < 8; ++j) { float f = bf2f(u.h[j]); s += f * f; }
  }
#pragma unroll
  for (int off = 16; off > 0; off >>= 1) s += __shfl_xor(s, off, 32);
  if (lane == 0) sq[row] = s;
}

// ---------------- gram + distance + sigmoid, scatter to packed upper-tri ----------------
__global__ __launch_bounds__(256) void k_gram(const unsigned short* __restrict__ Hall,
                                              const float* __restrict__ sqall,
                                              const float* __restrict__ thrp,
                                              float* __restrict__ probs) {
  const int b = blockIdx.z;
  const int m0 = blockIdx.y * 128;
  const int n0 = blockIdx.x * 64;
  if (n0 + 64 <= m0) return;   // block entirely below diagonal (uniform)

  __shared__ __align__(16) unsigned short As[2][128 * 32];
  __shared__ __align__(16) unsigned short Bs[2][64 * 32];
  const unsigned short* Hg = Hall + (size_t)b * NN * HID;
  const int t = threadIdx.x;
  const int lane = t & 31, wave = t >> 5;
  const int wm = wave >> 2, wn = wave & 3;
  const int mrow = lane & 15, kh = lane >> 4;

  const int ar0 = t >> 2, ar1 = ar0 + 64, seg = (t & 3) * 8;
  const int br = t >> 2;

  auto stage = [&](int buf, int k0) {
    cp16(Hg + (size_t)(m0 + ar0) * HID + k0 + seg, &As[buf][ar0 * 32 + seg]);
    cp16(Hg + (size_t)(m0 + ar1) * HID + k0 + seg, &As[buf][ar1 * 32 + seg]);
    cp16(Hg + (size_t)(n0 + br) * HID + k0 + seg, &Bs[buf][br * 32 + seg]);
  };

  v8f acc[4];
#pragma unroll
  for (int s = 0; s < 4; ++s)
#pragma unroll
    for (int r = 0; r < 8; ++r) acc[s][r] = 0.0f;

  const int nsteps = HID >> 5;
  stage(0, 0);
  for (int i = 0; i < nsteps; ++i) {
    const int cur = i & 1;
    async_wait_all();
    __syncthreads();
    if (i + 1 < nsteps) stage(cur ^ 1, (i + 1) << 5);

    const unsigned short* Asb = As[cur];
    const unsigned short* Bsb = Bs[cur];
    v16bf bfrag = frag_b(Bsb + (wn * 16 + mrow) * 32 + kh * 8);
    v16bf a0 = frag_a(Asb + (wm * 64 + 0 + mrow) * 32 + kh * 8);
    v16bf a1 = frag_a(Asb + (wm * 64 + 16 + mrow) * 32 + kh * 8);
    v16bf a2 = frag_a(Asb + (wm * 64 + 32 + mrow) * 32 + kh * 8);
    v16bf a3 = frag_a(Asb + (wm * 64 + 48 + mrow) * 32 + kh * 8);
    acc[0] = __builtin_amdgcn_wmma_f32_16x16x32_bf16(false, a0, false, bfrag, (short)0, acc[0], false, false);
    acc[1] = __builtin_amdgcn_wmma_f32_16x16x32_bf16(false, a1, false, bfrag, (short)0, acc[1], false, false);
    acc[2] = __builtin_amdgcn_wmma_f32_16x16x32_bf16(false, a2, false, bfrag, (short)0, acc[2], false, false);
    acc[3] = __builtin_amdgcn_wmma_f32_16x16x32_bf16(false, a3, false, bfrag, (short)0, acc[3], false, false);
  }

  const float thr = *thrp;
  const float* sqb = sqall + (size_t)b * NN;
  float* pb = probs + (size_t)b * PP;
  const int n = n0 + wn * 16 + mrow;
  const float sqn = sqb[n];
#pragma unroll
  for (int s = 0; s < 4; ++s) {
#pragma unroll
    for (int r = 0; r < 8; ++r) {
      int m = m0 + wm * 64 + s * 16 + r + kh * 8;
      if (m < n) {
        float d2 = sqb[m] + sqn - 2.0f * acc[s][r];
        d2 = fmaxf(d2, 1e-12f);
        float dist = sqrtf(d2);
        float prob = 1.0f / (1.0f + __expf(dist - thr));
        int pair = m * NN - (m * (m + 1)) / 2 + (n - m - 1);
        pb[pair] = prob;
      }
    }
  }
}

// ---------------- top-k: radix histogram over float bits ----------------
__global__ void k_hist(const float* __restrict__ probs, unsigned* __restrict__ bin_start,
                       unsigned* __restrict__ cursor) {
  const int b = blockIdx.x;
  __shared__ unsigned hist[NBINS];
  __shared__ unsigned start_s[NBINS];
  for (int i = threadIdx.x; i < NBINS; i += 256) hist[i] = 0;
  __syncthreads();
  const float* pb = probs + (size_t)b * PP;
  for (int i = threadIdx.x; i < PP; i += 256) {
    unsigned key = __float_as_uint(pb[i]);   // probs > 0 => bit-monotonic
    atomicAdd(&hist[key >> 21], 1u);
  }
  __syncthreads();
  if (threadIdx.x == 0) {
    unsigned cum = 0;
    for (int bin = NBINS - 1; bin >= 0; --bin) { start_s[bin] = cum; cum += hist[bin]; }
  }
  __syncthreads();
  for (int i = threadIdx.x; i < NBINS; i += 256) {
    bin_start[b * NBINS + i] = start_s[i];
    cursor[b * NBINS + i] = 0;
  }
}

__global__ void k_select(const float* __restrict__ probs, const unsigned* __restrict__ bin_start,
                         unsigned* __restrict__ cursor, int* __restrict__ idx_sel,
                         float* __restrict__ outP) {
  const int b = blockIdx.y;
  const int i = blockIdx.x * 256 + threadIdx.x;
  if (i >= PP) return;
  float p = probs[(size_t)b * PP + i];
  unsigned bin = __float_as_uint(p) >> 21;
  unsigned s = bin_start[b * NBINS + bin];
  if (s >= (unsigned)KSEL) return;                 // whole bin past K
  unsigned pos = s + atomicAdd(&cursor[b * NBINS + bin], 1u);
  if (pos < (unsigned)KSEL) {
    idx_sel[(size_t)b * KSEL + pos] = i;
    outP[(size_t)b * KSEL + pos] = p;
  }
}

// ---------------- edge_index assembly ----------------
__device__ __forceinline__ int tri_off(int u) { return u * NN - (u * (u + 1)) / 2; }

__global__ void k_edges(const int* __restrict__ idx_sel, int* __restrict__ outE) {
  int tid = blockIdx.x * 256 + threadIdx.x;
  if (tid >= NB * KSEL) return;
  int b = tid / KSEL, j = tid % KSEL;
  int i = idx_sel[tid];
  double disc = 767.0 * 767.0 - 8.0 * (double)i;   // 2N-1 = 767
  int u = (int)((767.0 - sqrt(disc)) * 0.5);
  if (u < 0) u = 0;
  if (u > NN - 2) u = NN - 2;
  while (u > 0 && tri_off(u) > i) --u;
  while (tri_off(u + 1) <= i) ++u;
  int v = i - tri_off(u) + u + 1;
  int uu = u + b * NN, vv = v + b * NN;
  const int E = NB * 2 * KSEL;
  outE[b * 2 * KSEL + j] = uu;          // row0: u then v
  outE[b * 2 * KSEL + KSEL + j] = vv;
  outE[E + b * 2 * KSEL + j] = vv;      // row1: v then u
  outE[E + b * 2 * KSEL + KSEL + j] = uu;
}

__global__ void k_batch(int* __restrict__ outB) {
  int i = blockIdx.x * 256 + threadIdx.x;
  if (i < ROWS) outB[i] = i / NN;
}

// ---------------- launcher ----------------
extern "C" void kernel_launch(void* const* d_in, const int* in_sizes, int n_in,
                              void* d_out, int out_size, void* d_ws, size_t ws_size,
                              hipStream_t stream) {
  const float* z      = (const float*)d_in[0];
  const int*   labels = (const int*)d_in[1];
  const float* ET     = (const float*)d_in[2];
  const float* W1     = (const float*)d_in[3];
  const float* b1     = (const float*)d_in[4];
  const float* W2     = (const float*)d_in[5];
  const float* b2     = (const float*)d_in[6];
  const float* We     = (const float*)d_in[7];
  const float* be     = (const float*)d_in[8];
  const float* thr    = (const float*)d_in[9];

  // workspace carve (256B aligned)
  char* w = (char*)d_ws;
  auto carve = [&](size_t bytes) {
    char* p = w;
    w += (bytes + 255) & ~(size_t)255;
    return p;
  };
  unsigned short* A0  = (unsigned short*)carve((size_t)ROWS * INDIM * 2);
  unsigned short* W1T = (unsigned short*)carve((size_t)HID * INDIM * 2);
  unsigned short* W2T = (unsigned short*)carve((size_t)FEAT * HID * 2);
  unsigned short* WET = (unsigned short*)carve((size_t)HID * FEAT * 2);
  unsigned short* H1  = (unsigned short*)carve((size_t)ROWS * HID * 2);
  unsigned short* XB  = (unsigned short*)carve((size_t)ROWS * FEAT * 2);
  unsigned short* HB  = (unsigned short*)carve((size_t)ROWS * HID * 2);
  float*          SQ  = (float*)carve((size_t)ROWS * 4);
  float*          PR  = (float*)carve((size_t)NB * PP * 4);
  unsigned*       BS  = (unsigned*)carve((size_t)NB * NBINS * 4);
  unsigned*       CU  = (unsigned*)carve((size_t)NB * NBINS * 4);
  int*            IDX = (int*)carve((size_t)NB * KSEL * 4);

  float* outX = (float*)d_out;
  int*   outE = (int*)d_out + X_ELE;
  int*   outB = (int*)d_out + X_ELE + EDGE_ELE;
  float* outP = (float*)d_out + X_ELE + EDGE_ELE + BATCH_ELE;

  // 1. pack inputs + weights to bf16 (weights transposed to [N][K])
  k_build_a0<<<(ROWS * INDIM + 255) / 256, 256, 0, stream>>>(z, labels, ET, A0);
  k_wt<<<(INDIM * HID + 255) / 256, 256, 0, stream>>>(W1, W1T, INDIM, HID);
  k_wt<<<(HID * FEAT + 255) / 256, 256, 0, stream>>>(W2, W2T, HID, FEAT);
  k_wt<<<(FEAT * HID + 255) / 256, 256, 0, stream>>>(We, WET, FEAT, HID);

  // 2. GEMM1: H1 = relu(A0 @ W1 + b1)   [24576,512]
  k_gemm<<<dim3(HID / 64, ROWS / 128), 256, 0, stream>>>(A0, W1T, b1, nullptr, H1, HID, INDIM, 1);
  // 3. GEMM2: X = H1 @ W2 + b2          [24576,256]  (f32 to d_out + bf16 for next GEMM)
  k_gemm<<<dim3(FEAT / 64, ROWS / 128), 256, 0, stream>>>(H1, W2T, b2, outX, XB, FEAT, HID, 0);
  // 4. GEMM3: H = relu(X @ We + be)     [24576,512]
  k_gemm<<<dim3(HID / 64, ROWS / 128), 256, 0, stream>>>(XB, WET, be, nullptr, HB, HID, FEAT, 1);

  // 5. sq[m] = ||h_m||^2
  k_rowsq<<<ROWS / 8, 256, 0, stream>>>(HB, SQ);

  // 6. gram + distance + sigmoid, upper-tri packed probs
  k_gram<<<dim3(NN / 64, NN / 128, NB), 256, 0, stream>>>(HB, SQ, thr, PR);

  // 7. per-graph top-K selection
  k_hist<<<NB, 256, 0, stream>>>(PR, BS, CU);
  k_select<<<dim3((PP + 255) / 256, NB), 256, 0, stream>>>(PR, BS, CU, IDX, outP);

  // 8. edge_index + batch
  k_edges<<<(NB * KSEL + 255) / 256, 256, 0, stream>>>(IDX, outE);
  k_batch<<<(ROWS + 255) / 256, 256, 0, stream>>>(outB);
}
